// mHCFlashStructureLayer_13168369729909
// MI455X (gfx1250) — matrix operand order repaired and checked by reading, not compiled, not run on hardware
//
#include <hip/hip_runtime.h>
#include <hip/hip_bf16.h>

// ---------------------------------------------------------------------------
// mHC Flash-IPA structure layer for gfx1250 (MI455X), wave32 + WMMA bf16.
// B=4, L=1024, CS=384, N_EXP=4, H=8, CH=64, PQK=4, PV=8, CZ=128, RANK=2.
// ---------------------------------------------------------------------------

typedef __attribute__((ext_vector_type(16))) __bf16 v16bf;
typedef __attribute__((ext_vector_type(8)))  float  v8f;

#define WMMA_BF16 __builtin_amdgcn_wmma_f32_16x16x32_bf16

#define BB   4
#define LL   1024
#define CSs  384
#define NEXP 4
#define HH   8
#define CHH  64
#define PQKp 4
#define PVp  8
#define CZQq 32
#define FEATN 1280
#define BL   (BB*LL)      // 4096
#define BHh  (BB*HH)      // 32

// A-operand (16x32 bf16) local K index for element e (0..15) in lane-group g.
__device__ __forceinline__ int a_kloc(int e, int g) {
    return e + 16 * g;   // lanes0-15: {0..15}; lanes16-31: {16..31}
}

// LDS byte offset from a generic pointer to __shared__ (flat LDS addr low 32b).
__device__ __forceinline__ unsigned lds_off(const void* p) {
    return (unsigned)(uintptr_t)p;
}

// ---------------------------------------------------------------------------
// Operand packers: emit WMMA-fragment-ordered bf16 so the GEMM inner loop is
// pure 32B vector loads + WMMA (no per-element gathers on the hot path).
//   PA[mt][kc][lane][e] = A[mt*16 + lane%16][kc*32 + a_kloc(e, lane/16)]
//   PW[nt][kc][lane][e] = W[kc*32 + 16*(lane/16) + e][nt*16 + lane%16]
// ---------------------------------------------------------------------------
__global__ void packA_kernel(const float* __restrict__ A, __bf16* __restrict__ PA,
                             int M, int K) {
    int idx = blockIdx.x * blockDim.x + threadIdx.x;
    int kt = K >> 5;
    int total = (M >> 4) * kt * 32;
    if (idx >= total) return;
    int lane = idx & 31; int t = idx >> 5;
    int kc = t % kt; int mt = t / kt;
    int g = lane >> 4, ln = lane & 15;
    const float* arow = A + (size_t)(mt * 16 + ln) * K + kc * 32;
    __bf16* o = PA + (size_t)idx * 16;
#pragma unroll
    for (int e = 0; e < 16; ++e) o[e] = (__bf16)arow[a_kloc(e, g)];
}

__global__ void packW_kernel(const float* __restrict__ W, __bf16* __restrict__ PW,
                             int K, int N) {
    int idx = blockIdx.x * blockDim.x + threadIdx.x;
    int kt = K >> 5;
    int total = (N >> 4) * kt * 32;
    if (idx >= total) return;
    int lane = idx & 31; int t = idx >> 5;
    int kc = t % kt; int nt = t / kt;
    int g = lane >> 4, ln = lane & 15;
    __bf16* o = PW + (size_t)idx * 16;
#pragma unroll
    for (int e = 0; e < 16; ++e)
        o[e] = (__bf16)W[(size_t)(kc * 32 + (g << 4) + e) * N + nt * 16 + ln];
}

// ---------------------------------------------------------------------------
// Packed bf16-WMMA GEMM: C[M,N] = act(A@W (+bias) (+res)). One wave per tile.
// Inner loop: 2x v16bf contiguous loads + 1 WMMA.
// ---------------------------------------------------------------------------
template<bool BIAS, bool RELU, bool RES>
__global__ __launch_bounds__(32)
void gemm_pk_kernel(const __bf16* __restrict__ PA, const __bf16* __restrict__ PW,
                    const float* __restrict__ bias, const float* __restrict__ res,
                    float* __restrict__ C, int M, int N, int K) {
    int tiles_n = N >> 4;
    int mt = blockIdx.x / tiles_n;
    int nt = blockIdx.x % tiles_n;
    int lane = threadIdx.x & 31;
    int g = lane >> 4, ln = lane & 15;
    int kt = K >> 5;
    const v16bf* pa = (const v16bf*)PA + (size_t)mt * kt * 32 + lane;
    const v16bf* pw = (const v16bf*)PW + (size_t)nt * kt * 32 + lane;
    v8f acc = {};
    for (int kc = 0; kc < kt; ++kc) {
        v16bf a = pa[(size_t)kc * 32];
        v16bf b = pw[(size_t)kc * 32];
        acc = WMMA_BF16(false, a, false, b, (short)0, acc, false, false);
    }
#pragma unroll
    for (int r = 0; r < 8; ++r) {
        int m = mt * 16 + r + (g << 3);
        int n = nt * 16 + ln;
        float v = acc[r];
        if (BIAS) v += bias[n];
        if (RES)  v += res[(size_t)m * N + n];
        if (RELU) v = v > 0.f ? v : 0.f;
        C[(size_t)m * N + n] = v;
    }
}

// ---------------------------------------------------------------------------
// Per-(b,l): mHC gates (H_post, Sinkhorn row-sums) + rotation from quaternion.
// Exploits s_c == s (softmax(H_pre) sums to 1 over identical streams).
// ---------------------------------------------------------------------------
__global__ void gates_rot_kernel(const float* __restrict__ s, const float* __restrict__ quat,
                                 const float* __restrict__ b_post, const float* __restrict__ b_res,
                                 const float* __restrict__ W_post, const float* __restrict__ W_res,
                                 const float* __restrict__ alpha_p,
                                 float* __restrict__ gates, float* __restrict__ rot) {
    int bl = blockIdx.x * blockDim.x + threadIdx.x;
    if (bl >= BL) return;
    const float* sr = s + (size_t)bl * CSs;
    float ss = 0.f;
    for (int c = 0; c < CSs; ++c) ss += sr[c] * sr[c];
    float rms = rsqrtf(ss / (float)CSs + 1e-6f);  // mean over 1536 of replicated s^2

    float dpost[4] = {0.f, 0.f, 0.f, 0.f};
    float dres[16];
    for (int j = 0; j < 16; ++j) dres[j] = 0.f;
    for (int c = 0; c < CSs; ++c) {
        float sv = sr[c] * rms;
        for (int n = 0; n < NEXP; ++n) {
            int row = n * CSs + c;
            const float* wp = W_post + (size_t)row * 4;
            const float* wr = W_res + (size_t)row * 16;
#pragma unroll
            for (int j = 0; j < 4; ++j) dpost[j] += sv * wp[j];
#pragma unroll
            for (int j = 0; j < 16; ++j) dres[j] += sv * wr[j];
        }
    }
    float al = alpha_p[0];
    float Rm[16]; float mx = -1e30f;
    for (int j = 0; j < 16; ++j) { Rm[j] = b_res[j] + al * dres[j]; mx = fmaxf(mx, Rm[j]); }
    for (int j = 0; j < 16; ++j) Rm[j] = __expf(Rm[j] - mx);
    for (int it = 0; it < 20; ++it) {
        for (int m = 0; m < 4; ++m) {
            float sro = Rm[m*4] + Rm[m*4+1] + Rm[m*4+2] + Rm[m*4+3] + 1e-8f;
            for (int n = 0; n < 4; ++n) Rm[m*4+n] /= sro;
        }
        for (int n = 0; n < 4; ++n) {
            float sco = Rm[n] + Rm[4+n] + Rm[8+n] + Rm[12+n] + 1e-8f;
            for (int m = 0; m < 4; ++m) Rm[m*4+n] /= sco;
        }
    }
    for (int m = 0; m < 4; ++m)
        gates[bl*8 + m] = Rm[m*4] + Rm[m*4+1] + Rm[m*4+2] + Rm[m*4+3];
    for (int m = 0; m < 4; ++m)
        gates[bl*8 + 4 + m] = 2.f / (1.f + __expf(-(b_post[m] + al * dpost[m])));

    const float* qt = quat + (size_t)bl * 4;
    float qn2 = qt[0]*qt[0] + qt[1]*qt[1] + qt[2]*qt[2] + qt[3]*qt[3];
    float inv = rsqrtf(qn2 + 1e-8f);
    float w = qt[0]*inv, x = qt[1]*inv, y = qt[2]*inv, z = qt[3]*inv;
    float* Ro = rot + (size_t)bl * 9;
    Ro[0] = 1.f - 2.f*(y*y + z*z); Ro[1] = 2.f*(x*y - w*z); Ro[2] = 2.f*(x*z + w*y);
    Ro[3] = 2.f*(x*y + w*z); Ro[4] = 1.f - 2.f*(x*x + z*z); Ro[5] = 2.f*(y*z - w*x);
    Ro[6] = 2.f*(x*z - w*y); Ro[7] = 2.f*(y*z + w*x); Ro[8] = 1.f - 2.f*(x*x + y*y);
}

// ---------------------------------------------------------------------------
// zq/zk = z_factor_1 @ Wz_{q,k} : per-(b,l) (tiny K=128, N=8 matvecs).
// ---------------------------------------------------------------------------
__global__ void zproj_kernel(const float* __restrict__ z1, const float* __restrict__ Wzq,
                             const float* __restrict__ Wzk,
                             float* __restrict__ zq, float* __restrict__ zk) {
    int bl = blockIdx.x * blockDim.x + threadIdx.x;
    if (bl >= BL) return;
    for (int r = 0; r < 2; ++r) {
        const float* zr = z1 + ((size_t)bl * 2 + r) * 128;
        float aq[8] = {}, ak[8] = {};
        for (int z = 0; z < 128; ++z) {
            float zv = zr[z];
#pragma unroll
            for (int h = 0; h < 8; ++h) {
                aq[h] += zv * Wzq[z*8 + h];
                ak[h] += zv * Wzk[z*8 + h];
            }
        }
        for (int h = 0; h < 8; ++h) {
            zq[bl*16 + r*8 + h] = aq[h];
            zk[bl*16 + r*8 + h] = ak[h];
        }
    }
}

// ---------------------------------------------------------------------------
// Build fused attention operands (per b,l,h):
//   Fq[bh,l,96] = [q*wL/sqrt(CH), wL*zq, 2*c*gamma*qp_g, 0pad]    (bf16)
//   Fk[bh,l,96] = [k, zk, kp_g, 0pad]                             (bf16)
//   GT[bh,160,L] = [v | vp_g | z2 | 0pad]^T                       (bf16)
//   qns/kns = c*gamma * sum||p_g||^2                              (f32)
// ---------------------------------------------------------------------------
__global__ void featbuild_kernel(const float* __restrict__ q, const float* __restrict__ k,
                                 const float* __restrict__ v, const float* __restrict__ qp,
                                 const float* __restrict__ kp, const float* __restrict__ vp,
                                 const float* __restrict__ zq, const float* __restrict__ zk,
                                 const float* __restrict__ z2, const float* __restrict__ rot,
                                 const float* __restrict__ trans, const float* __restrict__ head_w,
                                 __bf16* __restrict__ Fq, __bf16* __restrict__ Fk,
                                 __bf16* __restrict__ GT, float* __restrict__ qns,
                                 float* __restrict__ kns) {
    int idx = blockIdx.x * blockDim.x + threadIdx.x;
    if (idx >= BL * HH) return;
    int h = idx & 7; int bl = idx >> 3;
    int b = bl >> 10; int l = bl & 1023;
    int bh = b * HH + h;

    const float wL = 0.5773502691896258f;         // sqrt(1/3)
    const float wC = 0.2357022603955158f;         // sqrt(2/(9*PQK))
    float hw = head_w[h];
    float gamma = log1pf(__expf(hw));             // softplus
    float cg = wL * wC * 0.5f * gamma;
    float s2 = 2.f * cg;
    const float* R = rot + (size_t)bl * 9;
    const float* tt = trans + (size_t)bl * 3;

    __bf16* fqr = Fq + ((size_t)bh * LL + l) * 96;
    __bf16* fkr = Fk + ((size_t)bh * LL + l) * 96;
    const float qsc = wL / 8.f;                   // 1/sqrt(CH)=1/8
    for (int c = 0; c < CHH; ++c) {
        fqr[c] = (__bf16)(q[(size_t)bl * 512 + h * 64 + c] * qsc);
        fkr[c] = (__bf16)(k[(size_t)bl * 512 + h * 64 + c]);
    }
    fqr[64] = (__bf16)(zq[bl*16 + h] * wL);
    fqr[65] = (__bf16)(zq[bl*16 + 8 + h] * wL);
    fkr[64] = (__bf16)(zk[bl*16 + h]);
    fkr[65] = (__bf16)(zk[bl*16 + 8 + h]);
    float qn = 0.f, kn = 0.f;
    for (int p = 0; p < PQKp; ++p) {
        float xq[3], xk[3];
        for (int d = 0; d < 3; ++d) {
            xq[d] = qp[(size_t)bl * 96 + (h*PQKp + p)*3 + d];
            xk[d] = kp[(size_t)bl * 96 + (h*PQKp + p)*3 + d];
        }
        for (int i = 0; i < 3; ++i) {
            float gq = R[i*3]*xq[0] + R[i*3+1]*xq[1] + R[i*3+2]*xq[2] + tt[i];
            float gk = R[i*3]*xk[0] + R[i*3+1]*xk[1] + R[i*3+2]*xk[2] + tt[i];
            qn += gq*gq; kn += gk*gk;
            fqr[66 + p*3 + i] = (__bf16)(s2 * gq);
            fkr[66 + p*3 + i] = (__bf16)gk;
        }
    }
    for (int c = 78; c < 96; ++c) { fqr[c] = (__bf16)0.f; fkr[c] = (__bf16)0.f; }
    qns[(size_t)bh * LL + l] = cg * qn;
    kns[(size_t)bh * LL + l] = cg * kn;

    __bf16* gt = GT + (size_t)bh * 160 * LL;
    for (int c = 0; c < CHH; ++c)
        gt[(size_t)c * LL + l] = (__bf16)v[(size_t)bl * 512 + h * 64 + c];
    for (int p = 0; p < PVp; ++p) {
        float x0 = vp[(size_t)bl * 192 + (h*PVp + p)*3 + 0];
        float x1 = vp[(size_t)bl * 192 + (h*PVp + p)*3 + 1];
        float x2 = vp[(size_t)bl * 192 + (h*PVp + p)*3 + 2];
        for (int i = 0; i < 3; ++i) {
            float gv = R[i*3]*x0 + R[i*3+1]*x1 + R[i*3+2]*x2 + tt[i];
            gt[(size_t)(64 + p*3 + i) * LL + l] = (__bf16)gv;
        }
    }
    for (int r = 0; r < 2; ++r)
        for (int c = 0; c < CZQq; ++c)
            gt[(size_t)(88 + r*CZQq + c) * LL + l] =
                (__bf16)z2[(((size_t)bl * HH + h) * 2 + r) * CZQq + c];
    for (int c = 152; c < 160; ++c) gt[(size_t)c * LL + l] = (__bf16)0.f;
}

// ---------------------------------------------------------------------------
// Flash attention: one wave per (b,h, 16-row strip). Fk 32-key block staged to
// LDS via GLOBAL_LOAD_ASYNC_TO_LDS_B128 (+ s_wait_asynccnt); logits via 6
// WMMAs (K=96) with B-fragments read from LDS; softmax P relayout C->A through
// LDS; outputs via 10 WMMAs over GT (160 cols) per block.
// ---------------------------------------------------------------------------
__global__ __launch_bounds__(128)
void attn_kernel(const __bf16* __restrict__ Fq, const __bf16* __restrict__ Fk,
                 const __bf16* __restrict__ GT, const float* __restrict__ qns,
                 const float* __restrict__ kns, const float* __restrict__ mask,
                 float* __restrict__ feat, float* __restrict__ optg) {
    __shared__ float ldsP[4][16 * 33];
    __shared__ __attribute__((aligned(32))) __bf16 shFk[4][32 * 96];  // 6KB/wave
    int wave = threadIdx.x >> 5;
    int lane = threadIdx.x & 31;
    int g = lane >> 4, ln = lane & 15;
    int strip = blockIdx.x * 4 + wave;          // 0..2047
    int i16 = strip & 63;
    int bh = strip >> 6;
    int b = bh >> 3, h = bh & 7;
    int m0 = i16 << 4;

    // resident Fq A-fragments (3 x K=32)
    v16bf fq[3];
    const __bf16* fqrow = Fq + ((size_t)bh * LL + m0 + ln) * 96;
#pragma unroll
    for (int c = 0; c < 3; ++c)
#pragma unroll
        for (int e = 0; e < 16; ++e)
            fq[c][e] = fqrow[c * 32 + a_kloc(e, g)];

    float qr[8];
#pragma unroll
    for (int r = 0; r < 8; ++r) qr[r] = qns[(size_t)bh * LL + m0 + r + (g << 3)];

    float mrow[8], lrow[8];
    v8f O[10];
#pragma unroll
    for (int r = 0; r < 8; ++r) { mrow[r] = -3.0e38f; lrow[r] = 0.f; }
#pragma unroll
    for (int t = 0; t < 10; ++t) O[t] = (v8f){};

    unsigned fkLds = lds_off(&shFk[wave][0]);
    const __bf16* fkBase = &shFk[wave][0];

    for (int j0 = 0; j0 < LL; j0 += 32) {
        // --- async-stage this 32-key Fk block (32 rows x 192B contiguous) ---
        {
            unsigned long long ga =
                (unsigned long long)(uintptr_t)(Fk + ((size_t)bh * LL + j0) * 96)
                + (unsigned)lane * 16u;
            unsigned la = fkLds + (unsigned)lane * 16u;
#pragma unroll
            for (int i = 0; i < 12; ++i) {   // 12 * 512B = 6144B
                asm volatile("global_load_async_to_lds_b128 %0, %1, off"
                             :: "v"(la + (unsigned)(i * 512)),
                                "v"(ga + (unsigned long long)(i * 512))
                             : "memory");
            }
            asm volatile("s_wait_asynccnt 0" ::: "memory");
            __builtin_amdgcn_wave_barrier();
        }
        if (j0 + 32 < LL)
            __builtin_prefetch((const void*)(Fk + ((size_t)bh * LL + j0 + 32 + ln) * 96), 0, 1);

        v8f S0 = {}, S1 = {};
#pragma unroll
        for (int c = 0; c < 3; ++c) {
            v16bf b0 = *(const v16bf*)(fkBase + (size_t)ln * 96 + c * 32 + (g << 4));
            v16bf b1 = *(const v16bf*)(fkBase + (size_t)(16 + ln) * 96 + c * 32 + (g << 4));
            S0 = WMMA_BF16(false, fq[c], false, b0, (short)0, S0, false, false);
            S1 = WMMA_BF16(false, fq[c], false, b1, (short)0, S1, false, false);
        }
        float kn0 = kns[(size_t)bh * LL + j0 + ln];
        float kn1 = kns[(size_t)bh * LL + j0 + 16 + ln];
        bool mk0 = mask[b * LL + j0 + ln] > 0.f;
        bool mk1 = mask[b * LL + j0 + 16 + ln] > 0.f;

        float alpha_[8];
#pragma unroll
        for (int r = 0; r < 8; ++r) {
            float s0 = mk0 ? (S0[r] - qr[r] - kn0) : -1e9f;
            float s1 = mk1 ? (S1[r] - qr[r] - kn1) : -1e9f;
            float mx = fmaxf(s0, s1);
#pragma unroll
            for (int d = 8; d >= 1; d >>= 1) mx = fmaxf(mx, __shfl_xor(mx, d, 32));
            float mnew = fmaxf(mrow[r], mx);
            alpha_[r] = __expf(mrow[r] - mnew);
            mrow[r] = mnew;
            float p0 = __expf(s0 - mnew);
            float p1 = __expf(s1 - mnew);
            S0[r] = p0; S1[r] = p1;
            float rs = p0 + p1;
#pragma unroll
            for (int d = 8; d >= 1; d >>= 1) rs += __shfl_xor(rs, d, 32);
            lrow[r] = lrow[r] * alpha_[r] + rs;
#pragma unroll
            for (int t = 0; t < 10; ++t) O[t][r] *= alpha_[r];
        }

        // C-layout P -> LDS -> A-layout bf16 fragment
        float* P = ldsP[wave];
#pragma unroll
        for (int r = 0; r < 8; ++r) {
            int row = r + (g << 3);
            P[row * 33 + ln] = S0[r];
            P[row * 33 + 16 + ln] = S1[r];
        }
        __builtin_amdgcn_wave_barrier();
        asm volatile("s_wait_dscnt 0" ::: "memory");
        v16bf pa;
#pragma unroll
        for (int e = 0; e < 16; ++e)
            pa[e] = (__bf16)P[ln * 33 + a_kloc(e, g)];
        __builtin_amdgcn_wave_barrier();

#pragma unroll
        for (int t = 0; t < 10; ++t) {
            const __bf16* gp = GT + ((size_t)bh * 160 + t * 16 + ln) * LL + j0 + (g << 4);
            v16bf gb = *(const v16bf*)gp;
            O[t] = WMMA_BF16(false, pa, false, gb, (short)0, O[t], false, false);
        }
    }

    // epilogue: divide by softmax denom, scatter to feat / optg
#pragma unroll
    for (int r = 0; r < 8; ++r) {
        float inv = 1.f / lrow[r];
        int row = m0 + r + (g << 3);
        size_t fbase = ((size_t)b * LL + row) * FEATN;
#pragma unroll
        for (int t = 0; t < 10; ++t) {
            int col = t * 16 + ln;
            float val = O[t][r] * inv;
            if (col < 64)
                feat[fbase + h * 64 + col] = val;                      // o
            else if (col < 88)
                optg[((size_t)bh * LL + row) * 24 + (col - 64)] = val; // opt_g raw
            else if (col < 152)
                feat[fbase + 768 + h * 64 + (col - 88)] = val;         // opair
        }
    }
}

// ---------------------------------------------------------------------------
// opt_l = R^T (opt_g - t), opt_norm; thread per (b,l,h,p).
// ---------------------------------------------------------------------------
__global__ void optl_kernel(const float* __restrict__ optg, const float* __restrict__ rot,
                            const float* __restrict__ trans, float* __restrict__ feat) {
    int idx = blockIdx.x * blockDim.x + threadIdx.x;
    if (idx >= BL * HH * PVp) return;
    int p = idx & 7; int t1 = idx >> 3;
    int h = t1 & 7; int bl = t1 >> 3;
    int b = bl >> 10; int l = bl & 1023;
    int bh = b * HH + h;
    const float* R = rot + (size_t)bl * 9;
    const float* tt = trans + (size_t)bl * 3;
    float d0 = optg[((size_t)bh * LL + l) * 24 + p*3 + 0] - tt[0];
    float d1 = optg[((size_t)bh * LL + l) * 24 + p*3 + 1] - tt[1];
    float d2 = optg[((size_t)bh * LL + l) * 24 + p*3 + 2] - tt[2];
    float nrm = 0.f;
    size_t fbase = (size_t)bl * FEATN;
    for (int j = 0; j < 3; ++j) {
        float ol = R[0*3 + j]*d0 + R[1*3 + j]*d1 + R[2*3 + j]*d2;  // R^T diff
        nrm += ol * ol;
        feat[fbase + 512 + (h*PVp + p)*3 + j] = ol;
    }
    feat[fbase + 704 + h*PVp + p] = sqrtf(nrm + 1e-8f);
}

// ---------------------------------------------------------------------------
// LayerNorm over CS=384, one wave per row.
// ---------------------------------------------------------------------------
__global__ __launch_bounds__(128)
void ln_kernel(const float* __restrict__ x, const float* __restrict__ gw,
               const float* __restrict__ bw, float* __restrict__ out) {
    int wave = threadIdx.x >> 5, lane = threadIdx.x & 31;
    int row = blockIdx.x * 4 + wave;
    if (row >= BL) return;
    const float* xr = x + (size_t)row * CSs;
    float vals[12], sm = 0.f;
#pragma unroll
    for (int i = 0; i < 12; ++i) { vals[i] = xr[lane + i * 32]; sm += vals[i]; }
#pragma unroll
    for (int d = 16; d >= 1; d >>= 1) sm += __shfl_xor(sm, d, 32);
    float mu = sm / (float)CSs;
    float vs = 0.f;
#pragma unroll
    for (int i = 0; i < 12; ++i) { float dd = vals[i] - mu; vs += dd * dd; }
#pragma unroll
    for (int d = 16; d >= 1; d >>= 1) vs += __shfl_xor(vs, d, 32);
    float inv = rsqrtf(vs / (float)CSs + 1e-5f);
#pragma unroll
    for (int i = 0; i < 12; ++i) {
        int c = lane + i * 32;
        out[(size_t)row * CSs + c] = (vals[i] - mu) * inv * gw[c] + bw[c];
    }
}

// ---------------------------------------------------------------------------
// Backbone update: upd = s_ipa @ Wbb + bbb; compose frames. Thread per (b,l).
// ---------------------------------------------------------------------------
__global__ void bb_kernel(const float* __restrict__ s_ipa, const float* __restrict__ Wbb,
                          const float* __restrict__ bbb, const float* __restrict__ rot,
                          const float* __restrict__ trans,
                          float* __restrict__ out_rot, float* __restrict__ out_trans) {
    int bl = blockIdx.x * blockDim.x + threadIdx.x;
    if (bl >= BL) return;
    float u[6];
    const float* sr = s_ipa + (size_t)bl * CSs;
    for (int j = 0; j < 6; ++j) u[j] = bbb[j];
    for (int c = 0; c < CSs; ++c) {
        float sv = sr[c];
#pragma unroll
        for (int j = 0; j < 6; ++j) u[j] += sv * Wbb[c*6 + j];
    }
    float inv = rsqrtf(1.f + u[0]*u[0] + u[1]*u[1] + u[2]*u[2] + 1e-8f);
    float w = inv, x = u[0]*inv, y = u[1]*inv, z = u[2]*inv;
    float Ru[9];
    Ru[0] = 1.f - 2.f*(y*y + z*z); Ru[1] = 2.f*(x*y - w*z); Ru[2] = 2.f*(x*z + w*y);
    Ru[3] = 2.f*(x*y + w*z); Ru[4] = 1.f - 2.f*(x*x + z*z); Ru[5] = 2.f*(y*z - w*x);
    Ru[6] = 2.f*(x*z - w*y); Ru[7] = 2.f*(y*z + w*x); Ru[8] = 1.f - 2.f*(x*x + y*y);
    const float* R = rot + (size_t)bl * 9;
    float* Ro = out_rot + (size_t)bl * 9;
    for (int i = 0; i < 3; ++i)
        for (int j = 0; j < 3; ++j)
            Ro[i*3 + j] = R[i*3]*Ru[j] + R[i*3+1]*Ru[3 + j] + R[i*3+2]*Ru[6 + j];
    const float* tt = trans + (size_t)bl * 3;
    for (int i = 0; i < 3; ++i)
        out_trans[(size_t)bl*3 + i] = R[i*3]*u[3] + R[i*3+1]*u[4] + R[i*3+2]*u[5] + tt[i];
}

// ---------------------------------------------------------------------------
// mHC combine: s_out[b,l,m,:] = rowsum(H_res)[m]*s + H_post[m]*s_ipa.
// ---------------------------------------------------------------------------
__global__ void combine_kernel(const float* __restrict__ s, const float* __restrict__ s_ipa,
                               const float* __restrict__ gates, float* __restrict__ out) {
    size_t i = (size_t)blockIdx.x * blockDim.x + threadIdx.x;
    if (i >= (size_t)BL * NEXP * CSs) return;
    int c = (int)(i % CSs);
    int m = (int)((i / CSs) % NEXP);
    size_t bl = i / ((size_t)CSs * NEXP);
    out[i] = gates[bl*8 + m] * s[bl*CSs + c] + gates[bl*8 + 4 + m] * s_ipa[bl*CSs + c];
}

// ---------------------------------------------------------------------------
// Host launcher
// ---------------------------------------------------------------------------
extern "C" void kernel_launch(void* const* d_in, const int* in_sizes, int n_in,
                              void* d_out, int out_size, void* d_ws, size_t ws_size,
                              hipStream_t stream) {
    const float* s      = (const float*)d_in[0];
    const float* z1     = (const float*)d_in[1];
    const float* z2     = (const float*)d_in[2];
    const float* quat   = (const float*)d_in[3];
    const float* trans  = (const float*)d_in[4];
    const float* maskp  = (const float*)d_in[5];
    const float* b_post = (const float*)d_in[7];
    const float* b_res  = (const float*)d_in[8];
    const float* W_post = (const float*)d_in[10];
    const float* W_res  = (const float*)d_in[11];
    const float* alphap = (const float*)d_in[12];
    const float* Wq     = (const float*)d_in[13];
    const float* Wk     = (const float*)d_in[14];
    const float* Wv     = (const float*)d_in[15];
    const float* Wq_pts = (const float*)d_in[16];
    const float* Wk_pts = (const float*)d_in[17];
    const float* Wv_pts = (const float*)d_in[18];
    const float* Wz_q   = (const float*)d_in[19];
    const float* Wz_k   = (const float*)d_in[20];
    const float* head_w = (const float*)d_in[21];
    const float* Wo     = (const float*)d_in[22];
    const float* bo     = (const float*)d_in[23];
    const float* ln1_g  = (const float*)d_in[24];
    const float* ln1_b  = (const float*)d_in[25];
    const float* Wt1    = (const float*)d_in[26];
    const float* bt1    = (const float*)d_in[27];
    const float* Wt2    = (const float*)d_in[28];
    const float* bt2    = (const float*)d_in[29];
    const float* Wt3    = (const float*)d_in[30];
    const float* bt3    = (const float*)d_in[31];
    const float* lnt_g  = (const float*)d_in[32];
    const float* lnt_b  = (const float*)d_in[33];
    const float* Wbb    = (const float*)d_in[34];
    const float* bbb    = (const float*)d_in[35];

    // ---- workspace carve (256B aligned bump allocator) ----
    char* ws = (char*)d_ws;
    size_t off = 0;
    auto carve = [&](size_t bytes) -> void* {
        off = (off + 255) & ~(size_t)255;
        void* p = ws + off;
        off += bytes;
        return p;
    };
    float*  gates = (float*)carve((size_t)BL * 8 * 4);
    float*  rotb  = (float*)carve((size_t)BL * 9 * 4);
    float*  zqb   = (float*)carve((size_t)BL * 16 * 4);
    float*  zkb   = (float*)carve((size_t)BL * 16 * 4);
    float*  qns   = (float*)carve((size_t)BHh * LL * 4);
    float*  kns   = (float*)carve((size_t)BHh * LL * 4);
    float*  optg  = (float*)carve((size_t)BHh * LL * 24 * 4);
    __bf16* Fq    = (__bf16*)carve((size_t)BHh * LL * 96 * 2);
    __bf16* Fk    = (__bf16*)carve((size_t)BHh * LL * 96 * 2);
    __bf16* GT    = (__bf16*)carve((size_t)BHh * 160 * LL * 2);
    // region A: projections, later aliased by feat (feat <= proj footprint)
    char*  regA   = (char*)carve((size_t)BL * (512*3 + 96*2 + 192) * 4);
    float* qbuf   = (float*)regA;
    float* kbuf   = qbuf + (size_t)BL * 512;
    float* vbuf   = kbuf + (size_t)BL * 512;
    float* qpbuf  = vbuf + (size_t)BL * 512;
    float* kpbuf  = qpbuf + (size_t)BL * 96;
    float* vpbuf  = kpbuf + (size_t)BL * 96;
    float* feat   = (float*)regA;                  // alias (used after featbuild)
    float* bufA   = (float*)carve((size_t)BL * CSs * 4);   // x1 / t2 / s_ipa
    float* bufB   = (float*)carve((size_t)BL * CSs * 4);   // s1
    float* bufC   = (float*)carve((size_t)BL * CSs * 4);   // t1 / x2
    // packed WMMA operands
    __bf16* PA    = (__bf16*)carve((size_t)BL * FEATN * 2);       // activation pack scratch
    __bf16* PWq   = (__bf16*)carve((size_t)CSs * 512 * 2);
    __bf16* PWk   = (__bf16*)carve((size_t)CSs * 512 * 2);
    __bf16* PWv   = (__bf16*)carve((size_t)CSs * 512 * 2);
    __bf16* PWqp  = (__bf16*)carve((size_t)CSs * 96 * 2);
    __bf16* PWkp  = (__bf16*)carve((size_t)CSs * 96 * 2);
    __bf16* PWvp  = (__bf16*)carve((size_t)CSs * 192 * 2);
    __bf16* PWo   = (__bf16*)carve((size_t)FEATN * CSs * 2);
    __bf16* PWt1  = (__bf16*)carve((size_t)CSs * CSs * 2);
    __bf16* PWt2  = (__bf16*)carve((size_t)CSs * CSs * 2);
    __bf16* PWt3  = (__bf16*)carve((size_t)CSs * CSs * 2);

    float* out      = (float*)d_out;
    float* out_sout = out;                                  // [B,L,4,384]
    float* out_rot  = out + (size_t)BL * NEXP * CSs;        // [B,L,3,3]
    float* out_trn  = out_rot + (size_t)BL * 9;             // [B,L,3]

    auto packW = [&](const float* W, __bf16* PW, int K, int N) {
        int total = (N >> 4) * (K >> 5) * 32;
        packW_kernel<<<(total + 255) / 256, 256, 0, stream>>>(W, PW, K, N);
    };
    auto packA = [&](const float* A, int M, int K) {
        int total = (M >> 4) * (K >> 5) * 32;
        packA_kernel<<<(total + 255) / 256, 256, 0, stream>>>(A, PA, M, K);
    };

    // 0. pack all weight matrices into WMMA B-fragment order (bf16)
    packW(Wq, PWq, CSs, 512);  packW(Wk, PWk, CSs, 512);  packW(Wv, PWv, CSs, 512);
    packW(Wq_pts, PWqp, CSs, 96); packW(Wk_pts, PWkp, CSs, 96); packW(Wv_pts, PWvp, CSs, 192);
    packW(Wo, PWo, FEATN, CSs);
    packW(Wt1, PWt1, CSs, CSs); packW(Wt2, PWt2, CSs, CSs); packW(Wt3, PWt3, CSs, CSs);

    // 1. gates + rotation
    gates_rot_kernel<<<BL / 64, 64, 0, stream>>>(s, quat, b_post, b_res, W_post, W_res,
                                                 alphap, gates, rotb);
    // 2. projections (packed bf16 WMMA GEMMs), s_c == s
    packA(s, BL, CSs);
    gemm_pk_kernel<false,false,false><<<(BL/16)*(512/16), 32, 0, stream>>>(
        PA, PWq, nullptr, nullptr, qbuf, BL, 512, CSs);
    gemm_pk_kernel<false,false,false><<<(BL/16)*(512/16), 32, 0, stream>>>(
        PA, PWk, nullptr, nullptr, kbuf, BL, 512, CSs);
    gemm_pk_kernel<false,false,false><<<(BL/16)*(512/16), 32, 0, stream>>>(
        PA, PWv, nullptr, nullptr, vbuf, BL, 512, CSs);
    gemm_pk_kernel<false,false,false><<<(BL/16)*(96/16), 32, 0, stream>>>(
        PA, PWqp, nullptr, nullptr, qpbuf, BL, 96, CSs);
    gemm_pk_kernel<false,false,false><<<(BL/16)*(96/16), 32, 0, stream>>>(
        PA, PWkp, nullptr, nullptr, kpbuf, BL, 96, CSs);
    gemm_pk_kernel<false,false,false><<<(BL/16)*(192/16), 32, 0, stream>>>(
        PA, PWvp, nullptr, nullptr, vpbuf, BL, 192, CSs);
    // 3. pair-factor projections
    zproj_kernel<<<BL / 256, 256, 0, stream>>>(z1, Wz_q, Wz_k, zqb, zkb);
    // 4. fused attention operand build
    featbuild_kernel<<<(BL * HH) / 256, 256, 0, stream>>>(
        qbuf, kbuf, vbuf, qpbuf, kpbuf, vpbuf, zqb, zkb, z2, rotb, trans, head_w,
        Fq, Fk, GT, qns, kns);
    // 5. flash-IPA attention (writes o/opair into feat [aliases proj region], optg)
    attn_kernel<<<(BHh * (LL/16)) / 4, 128, 0, stream>>>(
        Fq, Fk, GT, qns, kns, maskp, feat, optg);
    // 6. local-frame point outputs
    optl_kernel<<<(BL * HH * PVp) / 256, 256, 0, stream>>>(optg, rotb, trans, feat);
    // 7. output projection + residual: x1 = s + feat@Wo + bo
    packA(feat, BL, FEATN);
    gemm_pk_kernel<true,false,true><<<(BL/16)*(CSs/16), 32, 0, stream>>>(
        PA, PWo, bo, s, bufA, BL, CSs, FEATN);
    // 8. s1 = LN(x1)
    ln_kernel<<<BL / 4, 128, 0, stream>>>(bufA, ln1_g, ln1_b, bufB);
    // 9-11. transition MLP with residual
    packA(bufB, BL, CSs);
    gemm_pk_kernel<true,true,false><<<(BL/16)*(CSs/16), 32, 0, stream>>>(
        PA, PWt1, bt1, nullptr, bufC, BL, CSs, CSs);
    packA(bufC, BL, CSs);
    gemm_pk_kernel<true,true,false><<<(BL/16)*(CSs/16), 32, 0, stream>>>(
        PA, PWt2, bt2, nullptr, bufA, BL, CSs, CSs);
    packA(bufA, BL, CSs);
    gemm_pk_kernel<true,false,true><<<(BL/16)*(CSs/16), 32, 0, stream>>>(
        PA, PWt3, bt3, bufB, bufC, BL, CSs, CSs);
    // 12. s_ipa = LN(x2)
    ln_kernel<<<BL / 4, 128, 0, stream>>>(bufC, lnt_g, lnt_b, bufA);
    // 13. backbone frame update -> d_out
    bb_kernel<<<BL / 64, 64, 0, stream>>>(bufA, Wbb, bbb, rotb, trans, out_rot, out_trn);
    // 14. mHC stream combine -> d_out
    combine_kernel<<<((size_t)BL * NEXP * CSs) / 256, 256, 0, stream>>>(
        s, bufA, gates, out_sout);
    (void)in_sizes; (void)n_in; (void)out_size; (void)ws_size;
}
